// EquivariantProductBasisBlock_28939489641124
// MI455X (gfx1250) — matrix operand order repaired and checked by reading, not compiled, not run on hardware
//
#include <hip/hip_runtime.h>

typedef __attribute__((ext_vector_type(16))) _Float16 v16h;
typedef __attribute__((ext_vector_type(8)))  float    v8f;

// Problem constants (from reference)
#define NNODES 1024
#define C      128
#define NELL   16
#define NE     10
#define K3     11     // paths nu=3
#define K3P    12     // padded
#define KDIM   176    // K3*NELL
#define KPAD   192    // padded to 6 k-tiles of 32
#define KT     6      // k-tiles of 32

// ---------------------------------------------------------------------------
// Prep A: element index from one-hot node_attrs
// ---------------------------------------------------------------------------
__global__ void prep_elem_kernel(const float* __restrict__ attrs, int* __restrict__ elem) {
    int b = blockIdx.x * blockDim.x + threadIdx.x;
    if (b >= NNODES) return;
    int best = 0; float bv = attrs[b * NE];
    for (int e = 1; e < NE; ++e) {
        float v = attrs[b * NE + e];
        if (v > bv) { bv = v; best = e; }
    }
    elem[b] = best;
}

// ---------------------------------------------------------------------------
// Prep B: pack U3 (f32 [d,16,16,16,11]) into f16 WMMA A-fragment layout.
// A tile is 16(M)x32(K) f16.  Per ISA 16-bit A layout:
//   lane L: M = L%16 ; half h (VGPR h/2, sel h%2):
//   K = ((h>=8)?16:0) + (L/16)*8 + ((h/2)%4)*2 + (h%2)
// Fragment stored so lane L reads its 16 halves contiguously at lane*16.
// li=0: 16 m-tiles (M=256);  li=1: 48 m-tiles (M=768).  6 k-tiles each.
// ---------------------------------------------------------------------------
__global__ void pack_u3_kernel(const float* __restrict__ U3_0,
                               const float* __restrict__ U3_1,
                               _Float16* __restrict__ apack) {
    int id = blockIdx.x * blockDim.x + threadIdx.x;     // < 384*512
    if (id >= 384 * 512) return;
    int frag   = id >> 9;
    int within = id & 511;
    int L = within >> 4;
    int h = within & 15;
    const float* U; _Float16* dst; int fl;
    if (frag < 96) { U = U3_0; fl = frag;      dst = apack; }
    else           { U = U3_1; fl = frag - 96; dst = apack + 96 * 512; }
    int mt = fl / KT, kt = fl % KT;
    int m  = mt * 16 + (L & 15);                         // m = (d*16+e1)*16+e2
    int kf = ((h >> 3) << 4) + ((L >> 4) << 3) + (((h >> 1) & 3) << 1) + (h & 1);
    int j  = kt * 32 + kf;                               // j = k*16 + i
    float v = 0.f;
    if (j < KDIM) v = U[m * KDIM + (j & 15) * K3 + (j >> 4)];
    dst[(size_t)fl * 512 + within] = (_Float16)v;
}

// ---------------------------------------------------------------------------
// Main fused kernel: one block = one node, one output irrep (li).
//  1) gather x[128][16], W3/W2/W1 rows for this node's element into LDS
//  2) per 16-channel column tile:
//       z[c,(k,i)] = x[c,i]*W3[e,k,c]  -> f16 B-fragments in LDS
//       out3[m,c]  = U3flat @ z        -> v_wmma_f32_16x16x32_f16 (f32 acc)
//       out2[d,e1,c] = sum_e2 (out3 + U2.W2)[.. ,e2] * x[c,e2]
//       feats[c,d]   = sum_e1 (out2 + U1.W1)[..,e1] * x[c,e1]
//  3) per-node channel mix: out[f,d] = lin^T @ feats  (WMMA), *1/sqrt(C), +sc
// ---------------------------------------------------------------------------
__global__ __launch_bounds__(256)
void mace_product_kernel(const float* __restrict__ node_feats,
                         const float* __restrict__ sc,
                         const float* __restrict__ U2,   // [d,16,16,4]
                         const float* __restrict__ U1,   // [d,16,1]
                         const float* __restrict__ W3,   // [E,11,C]
                         const float* __restrict__ W2,   // [E,4,C]
                         const float* __restrict__ W1,   // [E,1,C]
                         const float* __restrict__ lin,  // [C,C]
                         const int*   __restrict__ elem,
                         const _Float16* __restrict__ apack, // this li's fragments
                         float* __restrict__ out,
                         int d, int mtiles, int li) {
    __shared__ float xs[C * NELL];                  // 8 KB
    __shared__ float wb3s[K3P * C];                 // 6 KB (row 11 zeroed)
    __shared__ float wb2s[4 * C];                   // 2 KB
    __shared__ float wb1s[C];                       // 0.5 KB
    __shared__ alignas(32) _Float16 zfrag[KT * 512];    // 6 KB
    __shared__ float out3h[24 * 16 * 16];           // 24 KB (<=24 m-tiles/chunk)
    __shared__ float out2s[48 * 16];                // 3 KB
    __shared__ float featsL[C * 4];                 // 2 KB
    __shared__ alignas(32) _Float16 bfragL[4 * 512];    // 4 KB

    const int node = blockIdx.x;
    const int tid  = threadIdx.x;
    const int wave = tid >> 5;
    const int lane = tid & 31;
    const int e    = elem[node];

    if (lane == 0) __builtin_prefetch(apack + (size_t)wave * 6 * 512, 0, 1);

    // --- stage node data into LDS ---
    for (int t = tid; t < C * NELL; t += 256) xs[t] = node_feats[(size_t)node * C * NELL + t];
    for (int t = tid; t < K3P * C; t += 256) {
        int k = t >> 7, c = t & 127;
        wb3s[t] = (k < K3) ? W3[((size_t)e * K3 + k) * C + c] : 0.f;
    }
    for (int t = tid; t < 4 * C; t += 256) {
        int k = t >> 7, c = t & 127;
        wb2s[t] = W2[((size_t)e * 4 + k) * C + c];
    }
    for (int t = tid; t < C; t += 256) wb1s[t] = W1[(size_t)e * C + t];
    __syncthreads();

    const int nchunks = (mtiles + 23) / 24;

    for (int ct = 0; ct < 8; ++ct) {                // 8 column tiles of 16 channels
        // --- build B fragments: z[c, j=k*16+i] = x[c,i]*wb3[k,c] (f16) ---
        // B 16-bit layout (32xK x 16N): lane L: N=L%16, K=(h/8)*16+(L/16)*8+h%8
        for (int t = tid; t < KT * 512; t += 256) {
            int kt  = t >> 9;
            int rem = t & 511;
            int L   = rem >> 4;
            int h   = rem & 15;
            int c   = ct * 16 + (L & 15);
            int j   = kt * 32 + ((h >> 3) << 4) + ((L >> 4) << 3) + (h & 7);          // < 192; row 11 is zero
            float v = xs[c * NELL + (j & 15)] * wb3s[(j >> 4) * C + c];
            zfrag[t] = (_Float16)v;
        }
        __syncthreads();

        for (int chunk = 0; chunk < nchunks; ++chunk) {
            int cmt = mtiles - chunk * 24; if (cmt > 24) cmt = 24;
            // --- WMMA GEMM: out3[m, c] over K=192 ---
            for (int mtl = wave; mtl < cmt; mtl += 8) {
                int mtg = chunk * 24 + mtl;
                const _Float16* afr = apack + (size_t)mtg * KT * 512 + lane * 16;
                v8f acc = {};
                #pragma unroll
                for (int kt = 0; kt < KT; ++kt) {
                    v16h av = *(const v16h*)(afr + kt * 512);
                    v16h bv = *(const v16h*)&zfrag[kt * 512 + lane * 16];
                    acc = __builtin_amdgcn_wmma_f32_16x16x32_f16(
                              false, av, false, bv, (short)0, acc, false, false);
                }
                int cl = lane & 15;
                int rbase = mtl * 16 + ((lane >> 4) << 3);
                #pragma unroll
                for (int rr = 0; rr < 8; ++rr)
                    out3h[(rbase + rr) * 16 + cl] = acc[rr];
            }
            __syncthreads();
            // --- contract e2 with fused U2·W2 term (m-tile index == (d,e1)) ---
            int ntasks = cmt * 16;
            for (int t = tid; t < ntasks; t += 256) {
                int me1l = t >> 4, cl = t & 15;
                int me1  = chunk * 24 + me1l;
                int c    = ct * 16 + cl;
                float a2 = 0.f;
                for (int e2 = 0; e2 < 16; ++e2) {
                    const float* u2p = U2 + ((size_t)me1 * 16 + e2) * 4;
                    float u2e = 0.f;
                    #pragma unroll
                    for (int k = 0; k < 4; ++k) u2e += u2p[k] * wb2s[k * C + c];
                    a2 += (out3h[(me1l * 16 + e2) * 16 + cl] + u2e) * xs[c * NELL + e2];
                }
                out2s[me1 * 16 + cl] = a2;
            }
            __syncthreads();
        }
        // --- contract e1 with fused U1·W1 term -> feats[c,d] ---
        if (tid < 16 * d) {
            int dd = tid >> 4, cl = tid & 15;
            int c  = ct * 16 + cl;
            float f = 0.f;
            for (int e1 = 0; e1 < 16; ++e1)
                f += (out2s[(dd * 16 + e1) * 16 + cl] + U1[dd * 16 + e1] * wb1s[c])
                     * xs[c * NELL + e1];
            featsL[c * 4 + dd] = f;
        }
        __syncthreads();
    }

    // --- per-node channel-mixing linear via WMMA: out[f,n<d] = lin^T @ feats ---
    for (int t = tid; t < 4 * 512; t += 256) {
        int kt  = t >> 9;
        int rem = t & 511;
        int L   = rem >> 4;
        int h   = rem & 15;
        int cc  = kt * 32 + ((h >> 3) << 4) + ((L >> 4) << 3) + (h & 7);     // channel (K dim)
        int n   = L & 15;                            // column (d slot, padded)
        float v = (n < d) ? featsL[cc * 4 + n] : 0.f;
        bfragL[t] = (_Float16)v;
    }
    __syncthreads();
    {
        const int mt   = wave;                       // 8 f-tiles, 8 waves
        const int mrow = mt * 16 + (lane & 15);
        v8f acc = {};
        #pragma unroll
        for (int kt = 0; kt < 4; ++kt) {
            v16h av;
            #pragma unroll
            for (int h = 0; h < 16; ++h) {
                int ck = kt * 32 + ((h >> 3) << 4) + ((lane >> 4) << 3)
                       + (((h >> 1) & 3) << 1) + (h & 1);
                av[h] = (_Float16)lin[ck * C + mrow];   // A[f,c] = lin[c,f]
            }
            v16h bv = *(const v16h*)&bfragL[kt * 512 + lane * 16];
            acc = __builtin_amdgcn_wmma_f32_16x16x32_f16(
                      false, av, false, bv, (short)0, acc, false, false);
        }
        const float inv_sqrt_c = 0.08838834764831845f;  // 1/sqrt(128)
        int n = lane & 15;
        if (n < d) {
            #pragma unroll
            for (int rr = 0; rr < 8; ++rr) {
                int frow = mt * 16 + rr + ((lane >> 4) << 3);
                int col  = (li == 0) ? frow : (128 + frow * 3 + n);
                size_t idx = (size_t)node * 512 + col;
                out[idx] = acc[rr] * inv_sqrt_c + sc[idx];
            }
        }
    }
}

// ---------------------------------------------------------------------------
extern "C" void kernel_launch(void* const* d_in, const int* in_sizes, int n_in,
                              void* d_out, int out_size, void* d_ws, size_t ws_size,
                              hipStream_t stream) {
    const float* node_feats = (const float*)d_in[0];
    const float* sc         = (const float*)d_in[1];
    const float* attrs      = (const float*)d_in[2];
    const float* U01 = (const float*)d_in[3];
    const float* W01 = (const float*)d_in[4];
    const float* U02 = (const float*)d_in[5];
    const float* W02 = (const float*)d_in[6];
    const float* U03 = (const float*)d_in[7];
    const float* W03 = (const float*)d_in[8];
    const float* lin0 = (const float*)d_in[9];
    const float* U11 = (const float*)d_in[10];
    const float* W11 = (const float*)d_in[11];
    const float* U12 = (const float*)d_in[12];
    const float* W12 = (const float*)d_in[13];
    const float* U13 = (const float*)d_in[14];
    const float* W13 = (const float*)d_in[15];
    const float* lin1 = (const float*)d_in[16];
    float* out = (float*)d_out;

    // workspace: [0,4KB) elem idx; [4KB, +384KB) packed f16 U3 fragments
    char* ws = (char*)d_ws;
    int* elem = (int*)ws;
    _Float16* apack = (_Float16*)(ws + 4096);

    prep_elem_kernel<<<(NNODES + 255) / 256, 256, 0, stream>>>(attrs, elem);
    pack_u3_kernel<<<(384 * 512) / 256, 256, 0, stream>>>(U03, U13, apack);

    // li = 0: d=1, M=256 (16 m-tiles);  li = 1: d=3, M=768 (48 m-tiles)
    mace_product_kernel<<<NNODES, 256, 0, stream>>>(
        node_feats, sc, U02, U01, W03, W02, W01, lin0, elem,
        apack, out, 1, 16, 0);
    mace_product_kernel<<<NNODES, 256, 0, stream>>>(
        node_feats, sc, U12, U11, W13, W12, W11, lin1, elem,
        apack + 96 * 512, out, 3, 48, 1);
}
